// GCN_70411693851231
// MI455X (gfx1250) — matrix operand order repaired and checked by reading, not compiled, not run on hardware
//
#include <hip/hip_runtime.h>

typedef __bf16 bf16_t;
typedef __attribute__((ext_vector_type(16))) __bf16 v16bf;
typedef __attribute__((ext_vector_type(8)))  float  v8f;
typedef __attribute__((ext_vector_type(4)))  unsigned tdm_g0_t;
typedef __attribute__((ext_vector_type(8)))  unsigned tdm_g1_t;

#define N_NODES 50000
#define N_EDGES 1600000
#define NFEAT 256
#define NHID  64
#define NCLASS 16

// ---------------------------------------------------------------------------
// Pre-swizzle a row-major [K x N] fp32 weight matrix into per-lane bf16
// B-fragments matching the CDNA5 16-bit B (32x16) WMMA layout:
//   fragment (kb, nt): lane L holds column nt*16 + (L&15),
//   K values kb*32 + (L<16 ? 0..7,16..23 : 8..15,24..31), packed 2 per VGPR.
// ---------------------------------------------------------------------------
__global__ void prep_wfrag(const float* __restrict__ W, bf16_t* __restrict__ out,
                           int K, int N) {
    int ntiles = N >> 4;
    int total = (K >> 5) * ntiles * 32;
    int t = blockIdx.x * blockDim.x + threadIdx.x;
    if (t >= total) return;
    int L  = t & 31;
    int nt = (t >> 5) % ntiles;
    int kb = t / (32 * ntiles);
    int col   = nt * 16 + (L & 15);
    int kbase = kb * 32 + ((L < 16) ? 0 : 8);
    bf16_t* o = out + (size_t)t * 16;
#pragma unroll
    for (int i = 0; i < 16; ++i)
        o[i] = (bf16_t)W[(size_t)(kbase + (i < 8 ? i : i + 8)) * N + col];
}

// Convert 8 consecutive f32 to bf16 into a[off..off+7]
__device__ __forceinline__ void cvt8(const float* __restrict__ p, v16bf& a, int off) {
    float4 f0 = *(const float4*)(p);
    float4 f1 = *(const float4*)(p + 4);
    a[off + 0] = (bf16_t)f0.x; a[off + 1] = (bf16_t)f0.y;
    a[off + 2] = (bf16_t)f0.z; a[off + 3] = (bf16_t)f0.w;
    a[off + 4] = (bf16_t)f1.x; a[off + 5] = (bf16_t)f1.y;
    a[off + 6] = (bf16_t)f1.z; a[off + 7] = (bf16_t)f1.w;
}

// ---------------------------------------------------------------------------
// TDM: issue tensor_load_to_lds for a [rows x 32] f32 2-D tile of a row-major
// matrix with row stride NFEAT.  Descriptor per CDNA5 ISA ch.8 (2 SGPR groups,
// VADDR2/3 = NULL -> up-to-2D tensor).  tensor_dim1 = rows_remaining so OOB
// rows in the last block zero-fill instead of faulting.
// lds_addr is the ptrtoint of the actual __shared__ buffer (low 32 bits of a
// generic LDS pointer are the in-allocation byte address, ISA ch.10.2); this
// also CAPTURES the buffer so the compiler cannot prove the asm doesn't write
// it and must keep the subsequent ds_load fragment reads alive.
// ---------------------------------------------------------------------------
__device__ __forceinline__ void tdm_load_tile(const void* gptr, unsigned lds_addr,
                                              unsigned rows_rem) {
    unsigned long long ga = (unsigned long long)gptr;
    tdm_g0_t g0;
    g0[0] = 1u;                                            // count=1, user mode
    g0[1] = lds_addr;                                      // lds_addr (bytes)
    g0[2] = (unsigned)ga;                                  // global_addr[31:0]
    g0[3] = (unsigned)((ga >> 32) & 0x1FFFFFFu)            // global_addr[56:32]
          | (2u << 30);                                    // type=2 ("image")
    tdm_g1_t g1;
    g1[0] = (2u << 16);                                    // data_size=4B; mask=0
    g1[1] = (32u & 0xFFFFu) << 16;                         // tensor_dim0=32 (lo16)
    g1[2] = ((rows_rem & 0xFFFFu) << 16);                  // dim0 hi=0 | tensor_dim1 lo16
    g1[3] = ((rows_rem >> 16) & 0xFFFFu) | (32u << 16);    // tensor_dim1 hi | tile_dim0=32
    g1[4] = 128u;                                          // tile_dim1=128, tile_dim2=0
    g1[5] = (unsigned)NFEAT;                               // tensor_dim0_stride lo
    g1[6] = 0u;                                            // stride hi / dim1_stride lo
    g1[7] = 0u;
    asm volatile("tensor_load_to_lds %0, %1" :: "s"(g0), "s"(g1) : "memory");
}

// ---------------------------------------------------------------------------
// Layer-1 GEMM: C[50000 x 64] = A[50000 x 256] @ W1, bf16 WMMA / f32 accum.
// Block = 256 threads (8 waves) = 128 rows.  A tile (128x32 f32, 16 KB) is
// DMA'd into LDS by the Tensor Data Mover, double-buffered and overlapped
// with WMMA; fragments then come from LDS (ds_load_b128), B-fragments from
// the pre-swizzled global weight buffer.
// ---------------------------------------------------------------------------
__global__ __launch_bounds__(256)
void gemm1_tdm(const float* __restrict__ A, const bf16_t* __restrict__ Wfrag,
               float* __restrict__ C) {
    constexpr int KB = NFEAT / 32;     // 8 k-blocks
    constexpr int NT = NHID / 16;      // 4 n-tiles
    __shared__ float lbuf[2][128 * 32];   // 2 x 16 KB double buffer

    int tid  = threadIdx.x;
    int L    = tid & 31;
    int w    = tid >> 5;
    int row0 = blockIdx.x * 128;
    unsigned rows_rem = (unsigned)(N_NODES - row0);   // > 0 for all 391 blocks

    // In-allocation LDS byte addresses of the two buffers (captures lbuf).
    unsigned lds0 = (unsigned)(unsigned long long)(const void*)&lbuf[0][0];
    unsigned lds1 = (unsigned)(unsigned long long)(const void*)&lbuf[1][0];

    // Prologue: DMA k-block 0 into buffer 0 (wave 0 only; TDM ignores EXEC)
    if (tid < 32)
        tdm_load_tile(A + (size_t)row0 * NFEAT, lds0, rows_rem);

    int lrow = w * 16 + (L & 15);
    int koff = (L < 16) ? 0 : 8;        // A-matrix 16-bit layout half
    v8f acc[NT];
#pragma unroll
    for (int nt = 0; nt < NT; ++nt) acc[nt] = {};

    for (int kb = 0; kb < KB; ++kb) {
        if (tid < 32) __builtin_amdgcn_s_wait_tensorcnt(0);
        __syncthreads();                                  // tile kb ready for all
        if (kb + 1 < KB && tid < 32)                      // prefetch next tile
            tdm_load_tile(A + (size_t)row0 * NFEAT + (kb + 1) * 32,
                          ((kb + 1) & 1) ? lds1 : lds0, rows_rem);

        const float* lp = &lbuf[kb & 1][lrow * 32 + koff];
        v16bf a;
        cvt8(lp,      a, 0);                              // K = koff .. koff+7
        cvt8(lp + 16, a, 8);                              // K = 16+koff ..
#pragma unroll
        for (int nt = 0; nt < NT; ++nt) {
            v16bf b = *(const v16bf*)(Wfrag + (((size_t)kb * NT + nt) * 32 + L) * 16);
            acc[nt] = __builtin_amdgcn_wmma_f32_16x16x32_bf16(
                false, a, false, b, (short)0, acc[nt], false, false);
        }
        __syncthreads();                                  // all reads of buf done
    }

    // 32-bit C/D layout: VGPR v -> row tile0 + v (+8 for lanes 16..31), col=L&15
    int wtile = blockIdx.x * 8 + w;
    if (wtile < N_NODES / 16) {                           // mask surplus wave-tiles
        int rowoff = wtile * 16 + ((L < 16) ? 0 : 8);
        int col    = L & 15;
#pragma unroll
        for (int nt = 0; nt < NT; ++nt) {
#pragma unroll
            for (int v = 0; v < 8; ++v)
                C[(size_t)(rowoff + v) * NHID + nt * 16 + col] = acc[nt][v];
        }
    }
}

// ---------------------------------------------------------------------------
// Generic wave-tile GEMM (used for layer 2: K=64, N=16; weights tiny).
// ---------------------------------------------------------------------------
template <int K, int N>
__global__ __launch_bounds__(256)
void gemm_wmma(const float* __restrict__ A, const bf16_t* __restrict__ Wfrag,
               float* __restrict__ C) {
    constexpr int NT = N / 16;
    constexpr int KB = K / 32;
    int wave = (blockIdx.x * blockDim.x + threadIdx.x) >> 5;
    if (wave >= N_NODES / 16) return;          // wave-uniform early exit
    int L  = threadIdx.x & 31;
    int m0 = wave << 4;

    const float* arow = A + (size_t)(m0 + (L & 15)) * K;
    int koff = (L < 16) ? 0 : 8;

    v8f acc[NT];
#pragma unroll
    for (int nt = 0; nt < NT; ++nt) acc[nt] = {};

#pragma unroll
    for (int kb = 0; kb < KB; ++kb) {
        v16bf a;
        const float* p0 = arow + kb * 32 + koff;
        cvt8(p0,      a, 0);
        cvt8(p0 + 16, a, 8);
#pragma unroll
        for (int nt = 0; nt < NT; ++nt) {
            v16bf b = *(const v16bf*)(Wfrag + (((size_t)kb * NT + nt) * 32 + L) * 16);
            acc[nt] = __builtin_amdgcn_wmma_f32_16x16x32_bf16(
                false, a, false, b, (short)0, acc[nt], false, false);
        }
    }

    int rowoff = m0 + ((L < 16) ? 0 : 8);
    int col    = L & 15;
#pragma unroll
    for (int nt = 0; nt < NT; ++nt) {
#pragma unroll
        for (int v = 0; v < 8; ++v)
            C[(size_t)(rowoff + v) * N + nt * 16 + col] = acc[nt][v];
    }
}

// ---------------------------------------------------------------------------
// SPMM: out[dst] += w * sup[src]   (COO edges, float atomics; L2-resident)
// ---------------------------------------------------------------------------
template <int D>
__global__ __launch_bounds__(256)
void spmm_atomic(const int* __restrict__ src, const int* __restrict__ dst,
                 const float* __restrict__ w, const float* __restrict__ sup,
                 float* __restrict__ out) {
    constexpr int CH = D / 4;
    constexpr int SH = (CH == 16) ? 4 : 2;
    int t = blockIdx.x * blockDim.x + threadIdx.x;
    if (t >= N_EDGES * CH) return;
    int e = t >> SH;
    int c = t & (CH - 1);
    int s = src[e], d = dst[e];
    float wt = w[e];
    float4 v = *(const float4*)(sup + (size_t)s * D + c * 4);
    float* o = out + (size_t)d * D + c * 4;
    atomicAdd(o + 0, wt * v.x);
    atomicAdd(o + 1, wt * v.y);
    atomicAdd(o + 2, wt * v.z);
    atomicAdd(o + 3, wt * v.w);
}

__global__ void bias_relu(float* __restrict__ h, const float* __restrict__ b1) {
    int t = blockIdx.x * blockDim.x + threadIdx.x;
    if (t >= N_NODES * NHID) return;
    float v = h[t] + b1[t & (NHID - 1)];
    h[t] = v > 0.f ? v : 0.f;
}

__global__ void bias_logsoftmax(float* __restrict__ out, const float* __restrict__ b2) {
    int n = blockIdx.x * blockDim.x + threadIdx.x;
    if (n >= N_NODES) return;
    float* p = out + (size_t)n * NCLASS;
    float v[NCLASS];
    float mx = -3.0e38f;
#pragma unroll
    for (int i = 0; i < NCLASS; ++i) {
        v[i] = p[i] + b2[i];
        mx = v[i] > mx ? v[i] : mx;
    }
    float s = 0.f;
#pragma unroll
    for (int i = 0; i < NCLASS; ++i) s += __expf(v[i] - mx);
    float ls = mx + __logf(s);
#pragma unroll
    for (int i = 0; i < NCLASS; ++i) p[i] = v[i] - ls;
}

// ---------------------------------------------------------------------------
extern "C" void kernel_launch(void* const* d_in, const int* in_sizes, int n_in,
                              void* d_out, int out_size, void* d_ws, size_t ws_size,
                              hipStream_t stream) {
    const float* x   = (const float*)d_in[0];
    const float* W1  = (const float*)d_in[1];
    const float* b1  = (const float*)d_in[2];
    const float* W2  = (const float*)d_in[3];
    const float* b2  = (const float*)d_in[4];
    const int*   es  = (const int*)d_in[5];
    const int*   ed  = (const int*)d_in[6];
    const float* ew  = (const float*)d_in[7];
    float* out = (float*)d_out;

    // Workspace layout (all offsets 256B-aligned)
    char* base = (char*)d_ws;
    bf16_t* w1f  = (bf16_t*)(base);                       // 32 KB
    bf16_t* w2f  = (bf16_t*)(base + 40 * 1024);           //  2 KB
    float*  sup1 = (float*) (base + 64 * 1024);           // 12.8 MB
    float*  h    = sup1 + (size_t)N_NODES * NHID;         // 12.8 MB
    float*  sup2 = h    + (size_t)N_NODES * NHID;         // 3.2 MB

    // Weight pre-swizzle into WMMA B-fragment layout (bf16)
    prep_wfrag<<<4, 256, 0, stream>>>(W1, w1f, NFEAT, NHID);
    prep_wfrag<<<1, 64, 0, stream>>>(W2, w2f, NHID, NCLASS);

    // Layer 1: support1 = x @ W1  (TDM-staged A tiles, 128 rows/block)
    gemm1_tdm<<<391, 256, 0, stream>>>(x, w1f, sup1);

    // SPMM1: h[dst] += w * support1[src]
    hipMemsetAsync(h, 0, (size_t)N_NODES * NHID * sizeof(float), stream);
    spmm_atomic<NHID><<<(N_EDGES * 16) / 256, 256, 0, stream>>>(es, ed, ew, sup1, h);

    // h = relu(h + b1)
    bias_relu<<<(N_NODES * NHID) / 256, 256, 0, stream>>>(h, b1);

    // Layer 2: support2 = h @ W2
    gemm_wmma<NHID, NCLASS><<<391, 256, 0, stream>>>(h, w2f, sup2);

    // SPMM2 into d_out
    hipMemsetAsync(out, 0, (size_t)N_NODES * NCLASS * sizeof(float), stream);
    spmm_atomic<NCLASS><<<(N_EDGES * 4) / 256, 256, 0, stream>>>(es, ed, ew, sup2, out);

    // out = log_softmax(out + b2)
    bias_logsoftmax<<<(N_NODES + 255) / 256, 256, 0, stream>>>(out, b2);
}
